// Model_41102837023561
// MI455X (gfx1250) — matrix operand order repaired and checked by reading, not compiled, not run on hardware
//
#include <hip/hip_runtime.h>
#include <stdint.h>

// ---------------------------------------------------------------------------
// 3-layer LSTM (B=64,T=2048,D=40,H=164) + FC head for MI455X (gfx1250).
// Strategy:
//   * input projections -> big f16 WMMA GEMMs across the grid
//   * recurrence        -> 4 persistent workgroups (16 batch rows each),
//                          W_hh resident in LDS (264KB), h double-buffered,
//                          c/h state in VGPRs, TDM streams hs to global
// ---------------------------------------------------------------------------

#define TT   2048
#define DD   40
#define HH   164
#define NCH  11          // hidden chunks of 16 (H padded to 176)
#define NCOL 704         // 44 n-tiles * 16 (4H padded, gate-interleaved)
#define NTN  44          // n tiles
#define KPH  192         // padded K for hidden (6 k-tiles of 32)
#define NKTH 6

typedef __attribute__((ext_vector_type(16))) _Float16     v16h;
typedef __attribute__((ext_vector_type(8)))  float        v8f;
typedef __attribute__((ext_vector_type(4)))  unsigned int u32x4;
typedef __attribute__((ext_vector_type(8)))  int          i32x8;
typedef __attribute__((ext_vector_type(4)))  int          i32x4;

#if __has_builtin(__builtin_amdgcn_tensor_load_to_lds) && __has_builtin(__builtin_amdgcn_tensor_store_from_lds)
#define HAVE_TDM 1
#else
#define HAVE_TDM 0
#endif

union H16 { _Float16 h; unsigned short u; };

__device__ inline uint32_t lds_offset(const void* p) {
  return (uint32_t)(uintptr_t)p;   // LDS aperture: low 32 bits are the LDS byte offset
}

#if HAVE_TDM
// s_wait_tensorcnt requires a constant immediate
#define TDM_WAIT(n) __builtin_amdgcn_s_wait_tensorcnt(n)

// Build a 1D (flat, contiguous) TDM descriptor: data_size=8B, tile_dim0=n8.
__device__ inline void tdm_desc_1d(uint32_t lds, uint64_t gaddr, uint32_t nbytes,
                                   u32x4* g0, i32x8* g1) {
  uint32_t n8 = nbytes >> 3;
  (*g0)[0] = 1u;                                   // count=1, user mode
  (*g0)[1] = lds;                                  // lds_addr
  (*g0)[2] = (uint32_t)gaddr;                      // global_addr[31:0]
  (*g0)[3] = (uint32_t)(gaddr >> 32) | (2u << 30); // global_addr[56:32] | type=2
  (*g1)[0] = (int)(3u << 16);                      // wg_mask=0, data_size=3 (8B)
  (*g1)[1] = (int)((n8 & 0xFFFFu) << 16);          // tensor_dim0 low16 (bits 63:48)
  (*g1)[2] = (int)((n8 >> 16) & 0xFFFFu)           // tensor_dim0 high (bits 79:64)
           | (int)(1u << 16);                      // tensor_dim1 = 1
  (*g1)[3] = (int)((n8 & 0xFFFFu) << 16);          // tile_dim0 (bits 127:112)
  (*g1)[4] = 0;                                    // tile_dim1/2 unused
  (*g1)[5] = (int)n8;                              // tensor_dim0_stride low
  (*g1)[6] = 0; (*g1)[7] = 0;
}
__device__ inline void tdm_load_1d(uint32_t lds, uint64_t gaddr, uint32_t nbytes) {
  u32x4 g0; i32x8 g1; tdm_desc_1d(lds, gaddr, nbytes, &g0, &g1);
  i32x4 z4 = {0, 0, 0, 0};
  i32x8 z8 = {0, 0, 0, 0, 0, 0, 0, 0};
  __builtin_amdgcn_tensor_load_to_lds(g0, g1, z4, z4, z8, 0);
}
__device__ inline void tdm_store_1d(uint32_t lds, uint64_t gaddr, uint32_t nbytes) {
  u32x4 g0; i32x8 g1; tdm_desc_1d(lds, gaddr, nbytes, &g0, &g1);
  i32x4 z4 = {0, 0, 0, 0};
  i32x8 z8 = {0, 0, 0, 0, 0, 0, 0, 0};
  __builtin_amdgcn_tensor_store_from_lds(g0, g1, z4, z4, z8, 0);
}
#endif

// A-fragment (16x32 f16, M=batch rows) from an LDS row-major f16 buffer.
// lane L: M = L%16 ; VGPR v packs K pair kbase..kbase+1,
//   kbase = kt*32 + (L/16)*8 + (v<4 ? 2v : 16+2(v-4))
__device__ inline v16h load_afrag(const uint32_t* base32, int rowBase, int rowStrideDW,
                                  int kt, int lane) {
  union { uint32_t u[8]; v16h h; } a;
  const int row = rowBase + (lane & 15);
  const int grp = lane >> 4;
  const uint32_t* rp = base32 + (size_t)row * rowStrideDW + kt * 16 + grp * 4;
#pragma unroll
  for (int v = 0; v < 8; ++v) {
    const int kh = (v < 4) ? v : (8 + (v - 4));   // kbase/2 offset
    a.u[v] = rp[kh];
  }
  return a.h;
}

// B-fragment: prepacked in memory as tile-of-256-DWORDs, lane-major, 8 DW/lane.
__device__ inline v16h load_bfrag(const uint32_t* wlds, int tile, int lane) {
  union { uint32_t u[8]; v16h h; } b;
  const uint32_t* p = wlds + (size_t)tile * 256 + lane * 8;
#pragma unroll
  for (int v = 0; v < 8; ++v) b.u[v] = p[v];
  return b.h;
}

// ---------------------------------------------------------------------------
// Prep kernels
// ---------------------------------------------------------------------------

// x[B,T,D] f32 -> xp[T*64][64] f16 time-major, K zero-padded 40->64
__global__ void k_pad_x(const float* __restrict__ x, _Float16* __restrict__ xp) {
  const int total = TT * 64 * 64;
  int idx = blockIdx.x * blockDim.x + threadIdx.x;
  if (idx >= total) return;
  const int k = idx & 63, row = idx >> 6;
  const int b = row & 63, t = row >> 6;
  float v = (k < DD) ? x[((size_t)b * TT + t) * DD + k] : 0.f;
  xp[idx] = (_Float16)v;
}

// W[4H, kdim] f32 -> gate-interleaved, zero-padded B-fragment layout (f16 pairs)
// column n = hc*64 + gate*16 + col  <->  original row gate*164 + hc*16 + col
__global__ void k_wfrag(const float* __restrict__ W, int kdim, int nkt,
                        uint32_t* __restrict__ out) {
  const int total = nkt * NTN * 256;
  int idx = blockIdx.x * blockDim.x + threadIdx.x;
  if (idx >= total) return;
  const int v = idx & 7, lane = (idx >> 3) & 31, tile = idx >> 8;
  const int nt = tile / nkt, kt = tile % nkt;
  const int kb = kt * 32 + (lane >> 4) * 8 + (v < 4 ? 2 * v : 16 + 2 * (v - 4));
  uint32_t pk = 0;
#pragma unroll
  for (int d = 0; d < 2; ++d) {
    const int n = nt * 16 + (lane & 15); const int k = kb + d;
    const int hcq = n >> 6, g = (n >> 4) & 3, col = n & 15;
    const int jj = hcq * 16 + col;
    float val = (jj < HH && k < kdim) ? W[(size_t)(g * HH + jj) * kdim + k] : 0.f;
    H16 hv; hv.h = (_Float16)val;
    pk |= ((uint32_t)hv.u) << (16 * d);
  }
  out[idx] = pk;
}

// bias[4H] f32 -> reordered bias_r[704]
__global__ void k_bias(const float* __restrict__ b, float* __restrict__ out) {
  int idx = blockIdx.x * blockDim.x + threadIdx.x;
  if (idx >= NCOL) return;
  const int hcq = idx >> 6, g = (idx >> 4) & 3, col = idx & 15;
  const int jj = hcq * 16 + col;
  out[idx] = (jj < HH) ? b[g * HH + jj] : 0.f;
}

// ---------------------------------------------------------------------------
// Input-projection GEMM: gx[t-block] = A @ W_ih^T + bias, written in C-frag
// order (f16). One workgroup per t (64 rows), 11 waves (one hidden chunk each).
// ---------------------------------------------------------------------------
__launch_bounds__(352)
__global__ void k_gemm(const _Float16* __restrict__ A, int nkt,
                       const uint32_t* __restrict__ wih_frag,
                       const float* __restrict__ bias_r,
                       uint32_t* __restrict__ gx) {
  __shared__ __align__(16) uint32_t wlds[NKTH * NTN * 256];   // 270336 B (max)
  __shared__ __align__(16) uint32_t alds[64 * KPH / 2];        // 24576 B (max)
  const int tid = threadIdx.x, lane = tid & 31, hc = tid >> 5;
  const int t = blockIdx.x;
  const int Kp = nkt * 32;
  const int wDW = nkt * NTN * 256;
  const int aDW = 64 * Kp / 2;

#if HAVE_TDM
  if (tid < 32) {
    tdm_load_1d(lds_offset(wlds), (uint64_t)(uintptr_t)wih_frag, (uint32_t)wDW * 4u);
    tdm_load_1d(lds_offset(alds), (uint64_t)(uintptr_t)(A + (size_t)t * 64 * Kp),
                (uint32_t)(64 * Kp * 2));
    TDM_WAIT(0);
  }
#else
  for (int i = tid; i < wDW; i += 352) wlds[i] = wih_frag[i];
  {
    const uint32_t* ag = (const uint32_t*)(A + (size_t)t * 64 * Kp);
    for (int i = tid; i < aDW; i += 352) alds[i] = ag[i];
  }
#endif
  __syncthreads();

  float bv[4];
#pragma unroll
  for (int g = 0; g < 4; ++g) bv[g] = bias_r[hc * 64 + g * 16 + (lane & 15)];

  const v8f vz = {0.f, 0.f, 0.f, 0.f, 0.f, 0.f, 0.f, 0.f};
  for (int mt = 0; mt < 4; ++mt) {
    v8f acc[4];
#pragma unroll
    for (int g = 0; g < 4; ++g) acc[g] = vz;
    for (int kt = 0; kt < nkt; ++kt) {
      v16h a = load_afrag(alds, mt * 16, nkt * 16, kt, lane);
#pragma unroll
      for (int g = 0; g < 4; ++g) {
        v16h bfr = load_bfrag(wlds, (hc * 4 + g) * nkt + kt, lane);
        acc[g] = __builtin_amdgcn_wmma_f32_16x16x32_f16(
            false, a, false, bfr, (short)0, acc[g], false, false);
      }
    }
#pragma unroll
    for (int g = 0; g < 4; ++g) {
      union { u32x4 q; _Float16 h[8]; } ou;
#pragma unroll
      for (int r = 0; r < 8; ++r) ou.h[r] = (_Float16)(acc[g][r] + bv[g]);
      const size_t tI = (((size_t)t * 4 + mt) * NCH + hc) * 4 + g;
      *(u32x4*)(gx + tI * 128 + lane * 4) = ou.q;
    }
  }
}

// ---------------------------------------------------------------------------
// Recurrent scan: 4 workgroups (16 batch rows each), 11 waves, W_hh in LDS.
// Per step: gates = h @ W_hh^T (WMMA) + gx ; LSTM cell in VGPRs; masked by
// lengths; h double-buffered in LDS; hs streamed to global via TDM.
// ---------------------------------------------------------------------------
__launch_bounds__(352)
__global__ void k_scan(const uint32_t* __restrict__ gx,
                       const uint32_t* __restrict__ whh_frag,
                       const int* __restrict__ lengths,
                       _Float16* __restrict__ hs_out) {
  __shared__ __align__(16) uint32_t wlds[NKTH * NTN * 256];    // 270336 B
  __shared__ __align__(16) _Float16 hbuf[2][16 * KPH];         // 12288 B
  const int tid = threadIdx.x, lane = tid & 31, hc = tid >> 5;
  const int m = blockIdx.x;            // batch block (16 rows)
  const int grp = lane >> 4;
  const int j = hc * 16 + (lane & 15); // hidden unit owned by this lane

#if HAVE_TDM
  if (tid < 32)
    tdm_load_1d(lds_offset(wlds), (uint64_t)(uintptr_t)whh_frag,
                (uint32_t)(NKTH * NTN * 1024));
#else
  for (int i = tid; i < NKTH * NTN * 256; i += 352) wlds[i] = whh_frag[i];
#endif
  {
    _Float16* hb = &hbuf[0][0];
    for (int i = tid; i < 2 * 16 * KPH; i += 352) hb[i] = (_Float16)0.f;
  }
  int len8[8]; float c8[8], h8[8];
#pragma unroll
  for (int r = 0; r < 8; ++r) {
    len8[r] = lengths[m * 16 + grp * 8 + r];
    c8[r] = 0.f; h8[r] = 0.f;
  }
#if HAVE_TDM
  if (tid < 32) TDM_WAIT(0);
#endif
  __syncthreads();

  const v8f vz = {0.f, 0.f, 0.f, 0.f, 0.f, 0.f, 0.f, 0.f};
  int cur = 0;
  for (int t = 0; t < TT; ++t) {
#if HAVE_TDM
    if (tid < 32) TDM_WAIT(1);       // TDM store issued 2 steps ago is done
#endif
    __syncthreads();                 // release writes to buf[cur^1]

    v8f acc[4];
#pragma unroll
    for (int g = 0; g < 4; ++g) acc[g] = vz;
    const uint32_t* hb32 = (const uint32_t*)&hbuf[cur][0];
#pragma unroll
    for (int kt = 0; kt < NKTH; ++kt) {
      v16h a = load_afrag(hb32, 0, KPH / 2, kt, lane);
#pragma unroll
      for (int g = 0; g < 4; ++g) {
        v16h bfr = load_bfrag(wlds, (hc * 4 + g) * NKTH + kt, lane);
        acc[g] = __builtin_amdgcn_wmma_f32_16x16x32_f16(
            false, a, false, bfr, (short)0, acc[g], false, false);
      }
    }
    // add precomputed input projection + bias (C-fragment layout)
#pragma unroll
    for (int g = 0; g < 4; ++g) {
      const size_t tI = (((size_t)t * 4 + m) * NCH + hc) * 4 + g;
      union { u32x4 q; _Float16 h[8]; } gu;
      gu.q = *(const u32x4*)(gx + tI * 128 + lane * 4);
#pragma unroll
      for (int r = 0; r < 8; ++r) acc[g][r] += (float)gu.h[r];
    }
    // LSTM cell (gate order i,f,g,o), packed-sequence freeze
    _Float16* hn_out = &hbuf[cur ^ 1][0];
#pragma unroll
    for (int r = 0; r < 8; ++r) {
      const float si = 1.f / (1.f + __expf(-acc[0][r]));
      const float sf = 1.f / (1.f + __expf(-acc[1][r]));
      const float tg = tanhf(acc[2][r]);
      const float so = 1.f / (1.f + __expf(-acc[3][r]));
      float cn = sf * c8[r] + si * tg;
      float hn = so * tanhf(cn);
      if (t >= len8[r]) { cn = c8[r]; hn = h8[r]; }
      c8[r] = cn; h8[r] = hn;
      const float hw = (j < HH) ? hn : 0.f;
      hn_out[(grp * 8 + r) * KPH + j] = (_Float16)hw;
#if !HAVE_TDM
      hs_out[((size_t)t * 64 + m * 16 + grp * 8 + r) * KPH + j] = (_Float16)hw;
#endif
    }
#if !HAVE_TDM
    if (tid < 256) {   // zero pad columns 176..191 in global hs
      const int row = tid >> 4, col = 176 + (tid & 15);
      hs_out[((size_t)t * 64 + m * 16 + row) * KPH + col] = (_Float16)0.f;
    }
#endif
    __syncthreads();                 // all writes to buf[cur^1] done
#if HAVE_TDM
    if (tid < 32)
      tdm_store_1d(lds_offset(&hbuf[cur ^ 1][0]),
                   (uint64_t)(uintptr_t)hs_out + (size_t)t * 24576 + (size_t)m * 6144,
                   6144u);
#endif
    cur ^= 1;
  }
}

// ---------------------------------------------------------------------------
// FC head on the final hidden state (hs[T-1] == frozen h_T)
// ---------------------------------------------------------------------------
__global__ void k_fc(const _Float16* __restrict__ hT, const float* __restrict__ fw,
                     const float* __restrict__ fb, float* __restrict__ out) {
  const int tid = threadIdx.x;
  if (tid >= 64 * 7) return;
  const int b = tid / 7, o = tid % 7;
  float s = fb[o];
  for (int jj = 0; jj < HH; ++jj)
    s += (float)hT[(size_t)b * KPH + jj] * fw[o * HH + jj];
  out[b * 7 + o] = s;
}

// ---------------------------------------------------------------------------
extern "C" void kernel_launch(void* const* d_in, const int* in_sizes, int n_in,
                              void* d_out, int out_size, void* d_ws, size_t ws_size,
                              hipStream_t stream) {
  const float* x       = (const float*)d_in[0];
  const int*   lengths = (const int*)d_in[1];
  const float* W_ih_0  = (const float*)d_in[2];
  const float* W_hh_0  = (const float*)d_in[3];
  const float* b_0     = (const float*)d_in[4];
  const float* W_ih_1  = (const float*)d_in[5];
  const float* W_hh_1  = (const float*)d_in[6];
  const float* b_1     = (const float*)d_in[7];
  const float* W_ih_2  = (const float*)d_in[8];
  const float* W_hh_2  = (const float*)d_in[9];
  const float* b_2     = (const float*)d_in[10];
  const float* fc_w    = (const float*)d_in[11];
  const float* fc_b    = (const float*)d_in[12];

  char* ws = (char*)d_ws;
  size_t off = 0;
  auto carve = [&](size_t bytes) -> char* {
    char* p = ws + off;
    off += (bytes + 255) & ~(size_t)255;
    return p;
  };
  _Float16* xp     = (_Float16*)carve((size_t)TT * 64 * 64 * 2);       // 16.8 MB
  uint32_t* wf_ih0 = (uint32_t*)carve((size_t)2 * NTN * 1024);
  uint32_t* wf_hh0 = (uint32_t*)carve((size_t)NKTH * NTN * 1024);
  uint32_t* wf_ih1 = (uint32_t*)carve((size_t)NKTH * NTN * 1024);
  uint32_t* wf_hh1 = (uint32_t*)carve((size_t)NKTH * NTN * 1024);
  uint32_t* wf_ih2 = (uint32_t*)carve((size_t)NKTH * NTN * 1024);
  uint32_t* wf_hh2 = (uint32_t*)carve((size_t)NKTH * NTN * 1024);
  float*    br0    = (float*)carve(NCOL * 4);
  float*    br1    = (float*)carve(NCOL * 4);
  float*    br2    = (float*)carve(NCOL * 4);
  uint32_t* gx     = (uint32_t*)carve((size_t)TT * 64 * NCOL * 2);     // 184.5 MB
  _Float16* hs0    = (_Float16*)carve((size_t)TT * 64 * KPH * 2);      // 50.3 MB
  _Float16* hs1    = (_Float16*)carve((size_t)TT * 64 * KPH * 2);      // 50.3 MB
  (void)ws_size; (void)in_sizes; (void)n_in; (void)out_size;

  // ---- prep: pad/convert x, build fragment weights + reordered biases ----
  k_pad_x<<<(TT * 64 * 64 + 255) / 256, 256, 0, stream>>>(x, xp);
  k_wfrag<<<(2 * NTN * 256 + 255) / 256, 256, 0, stream>>>(W_ih_0, DD, 2, wf_ih0);
  k_wfrag<<<(NKTH * NTN * 256 + 255) / 256, 256, 0, stream>>>(W_hh_0, HH, NKTH, wf_hh0);
  k_wfrag<<<(NKTH * NTN * 256 + 255) / 256, 256, 0, stream>>>(W_ih_1, HH, NKTH, wf_ih1);
  k_wfrag<<<(NKTH * NTN * 256 + 255) / 256, 256, 0, stream>>>(W_hh_1, HH, NKTH, wf_hh1);
  k_wfrag<<<(NKTH * NTN * 256 + 255) / 256, 256, 0, stream>>>(W_ih_2, HH, NKTH, wf_ih2);
  k_wfrag<<<(NKTH * NTN * 256 + 255) / 256, 256, 0, stream>>>(W_hh_2, HH, NKTH, wf_hh2);
  k_bias<<<(NCOL + 255) / 256, 256, 0, stream>>>(b_0, br0);
  k_bias<<<(NCOL + 255) / 256, 256, 0, stream>>>(b_1, br1);
  k_bias<<<(NCOL + 255) / 256, 256, 0, stream>>>(b_2, br2);

  // ---- layer 0 ----
  k_gemm<<<TT, 352, 0, stream>>>(xp, 2, wf_ih0, br0, gx);
  k_scan<<<4, 352, 0, stream>>>(gx, wf_hh0, lengths, hs0);
  // ---- layer 1 ----
  k_gemm<<<TT, 352, 0, stream>>>(hs0, NKTH, wf_ih1, br1, gx);
  k_scan<<<4, 352, 0, stream>>>(gx, wf_hh1, lengths, hs1);
  // ---- layer 2 (reuse hs0 as hs2) ----
  k_gemm<<<TT, 352, 0, stream>>>(hs1, NKTH, wf_ih2, br2, gx);
  k_scan<<<4, 352, 0, stream>>>(gx, wf_hh2, lengths, hs0);

  // ---- FC on final hidden state ----
  k_fc<<<1, 448, 0, stream>>>(hs0 + (size_t)(TT - 1) * 64 * KPH, fc_w, fc_b,
                              (float*)d_out);
}
